// multihead_self_attention_22256520528886
// MI455X (gfx1250) — matrix operand order repaired and checked by reading, not compiled, not run on hardware
//
#include <hip/hip_runtime.h>
#include <cstdint>

// MI455X / gfx1250 multi-head causal self-attention.
// fp16 WMMA (V_WMMA_F32_16X16X32_F16) with fp32 accumulation throughout.
// Flash-attention inner loop; shared operands double-buffered in LDS via
// GLOBAL_LOAD_ASYNC_TO_LDS_B128 (ASYNCcnt) so the DMA overlaps the WMMAs.

typedef __attribute__((ext_vector_type(16))) _Float16 v16h;
typedef __attribute__((ext_vector_type(8)))  _Float16 v8h;
typedef __attribute__((ext_vector_type(8)))  float    v8f;

#define SEQ    4096
#define DMODEL 1024
#define HEADS  16
#define DK     64

// ---- async global -> LDS (CDNA5, tracked by ASYNCcnt) ---------------------
__device__ __forceinline__ void async_ld_b128(void* lds_ptr, const void* gptr) {
  // low 32 bits of a flat LDS pointer == LDS byte offset (ISA 10.2 aperture rule)
  uint32_t lds_off = (uint32_t)(uintptr_t)lds_ptr;
  asm volatile("global_load_async_to_lds_b128 %0, %1, off"
               :: "v"(lds_off), "v"(gptr) : "memory");
}
// Async loads complete in order: waiting for <=N leaves the newest N in flight.
__device__ __forceinline__ void wait_async0() {
  asm volatile("s_wait_asynccnt 0" ::: "memory");
}
__device__ __forceinline__ void wait_async1() {
  asm volatile("s_wait_asynccnt 1" ::: "memory");
}
__device__ __forceinline__ void wait_async2() {
  asm volatile("s_wait_asynccnt 2" ::: "memory");
}

// ---- WMMA operand loaders -------------------------------------------------
// A-matrix (16x32 f16): lane L holds row M=L%16; hf=L/16:
//   elements e = K = hf*16+e (0..15)
__device__ __forceinline__ v16h load_a_tile(const _Float16* row, int k0, int hf) {
  const v8h lo = *reinterpret_cast<const v8h*>(row + k0 + hf * 16);
  const v8h hi = *reinterpret_cast<const v8h*>(row + k0 + hf * 16 + 8);
  return __builtin_shufflevector(lo, hi, 0,1,2,3,4,5,6,7,8,9,10,11,12,13,14,15);
}

// B-matrix (32x16 f16): lane L holds col N=L%16, elements e = K = 16*(L/16)+e
// -> one contiguous 32B run per lane.
__device__ __forceinline__ v16h load_b16(const _Float16* p) {
  return *reinterpret_cast<const v16h*>(p);
}

__device__ __forceinline__ v8f wmma_f16(v16h a, v16h b, v8f c) {
  return __builtin_amdgcn_wmma_f32_16x16x32_f16(false, a, false, b, (short)0, c,
                                                false, false);
}

// ---- fp32 -> fp16 conversion ---------------------------------------------
__global__ void cvt_f32_f16_kernel(const float* __restrict__ in,
                                   _Float16* __restrict__ out, int n) {
  int i = blockIdx.x * blockDim.x + threadIdx.x;
  if (i < n) out[i] = (_Float16)in[i];
}

// ---- Fused QKV projection: q/k = xh @ W^T ; v stored transposed ----------
// grid (32, 16, 3), block 256 (8 waves). Wave: 16(M) x 64(N) output strip.
// The 64x32 f16 weight tile per K-step is double-buffered in LDS.
__global__ __launch_bounds__(256) void qkv_gemm_kernel(
    const _Float16* __restrict__ xh,
    const _Float16* __restrict__ wqh, const _Float16* __restrict__ wkh,
    const _Float16* __restrict__ wvh,
    _Float16* __restrict__ qh, _Float16* __restrict__ kh,
    _Float16* __restrict__ vt) {
  __shared__ _Float16 bsm[2][64 * 32];  // [buf][n_local][k_local] 2 x 4 KB
  const int t    = threadIdx.x;
  const int lane = t & 31;
  const int wave = t >> 5;
  const int hf   = lane >> 4;
  const int l16  = lane & 15;
  const int m0 = blockIdx.x * 128 + wave * 16;
  const int n0 = blockIdx.y * 64;
  const _Float16* w = (blockIdx.z == 0) ? wqh : (blockIdx.z == 1) ? wkh : wvh;

  const _Float16* arow = xh + (size_t)(m0 + l16) * DMODEL;

  // per-thread staging slot: 256 threads x 16B = 4 KB tile
  const int srow = t >> 2;            // 0..63  (weight row n0+srow)
  const int sch  = (t & 3) * 8;       // 0,8,16,24 (halfs within 32-K step)
  _Float16* sdst0 = bsm[0] + srow * 32 + sch;
  _Float16* sdst1 = bsm[1] + srow * 32 + sch;
  const _Float16* gsrc = w + (size_t)(n0 + srow) * DMODEL + sch;

  v8f acc0 = {}, acc1 = {}, acc2 = {}, acc3 = {};
  async_ld_b128(sdst0, gsrc);  // prologue fill of buffer 0
  for (int k0 = 0; k0 < DMODEL; k0 += 32) {
    const int cur = (k0 >> 5) & 1;
    const _Float16* bc = bsm[cur];
    if (k0 + 32 < DMODEL) {  // prefetch next tile into the other buffer
      async_ld_b128(cur ? sdst0 : sdst1, gsrc + k0 + 32);
      wait_async1();
    } else {
      wait_async0();
    }
    __syncthreads();
    v16h a  = load_a_tile(arow, k0, hf);
    v16h b0 = load_b16(bc + (l16 +  0) * 32 + hf * 16);
    v16h b1 = load_b16(bc + (l16 + 16) * 32 + hf * 16);
    v16h b2 = load_b16(bc + (l16 + 32) * 32 + hf * 16);
    v16h b3 = load_b16(bc + (l16 + 48) * 32 + hf * 16);
    acc0 = wmma_f16(a, b0, acc0);
    acc1 = wmma_f16(a, b1, acc1);
    acc2 = wmma_f16(a, b2, acc2);
    acc3 = wmma_f16(a, b3, acc3);
    __syncthreads();  // all reads of 'cur' done before its next overwrite
  }

  if (blockIdx.z < 2) {
    _Float16* out = (blockIdx.z == 0) ? qh : kh;
#pragma unroll
    for (int r = 0; r < 8; ++r) {
      const int m = m0 + r + 8 * hf;
      _Float16* orow = out + (size_t)m * DMODEL + n0 + l16;
      orow[0]  = (_Float16)acc0[r];
      orow[16] = (_Float16)acc1[r];
      orow[32] = (_Float16)acc2[r];
      orow[48] = (_Float16)acc3[r];
    }
  } else {  // V transposed: vt[n][m], n = head*64 + dk
#pragma unroll
    for (int r = 0; r < 8; ++r) {
      const int m = m0 + r + 8 * hf;
      vt[(size_t)(n0 + l16 +  0) * SEQ + m] = (_Float16)acc0[r];
      vt[(size_t)(n0 + l16 + 16) * SEQ + m] = (_Float16)acc1[r];
      vt[(size_t)(n0 + l16 + 32) * SEQ + m] = (_Float16)acc2[r];
      vt[(size_t)(n0 + l16 + 48) * SEQ + m] = (_Float16)acc3[r];
    }
  }
}

// ---- Flash attention ------------------------------------------------------
// grid 512, block 256 (8 waves). All 8 waves of a block handle the SAME head
// and 8 consecutive 16-query tiles; each 32-key step's K and V^T tiles are
// async-staged (double-buffered) in LDS and shared. Causal mask = -10000.
__global__ __launch_bounds__(256) void attn_kernel(
    const _Float16* __restrict__ qh, const _Float16* __restrict__ kh,
    const _Float16* __restrict__ vt, _Float16* __restrict__ ctxh) {
  __shared__ _Float16 Ksm[2][32 * 64];   // [buf][key_local][dk]  2 x 4 KB
  __shared__ _Float16 Vsm[2][64 * 32];   // [buf][dk][key_local]  2 x 4 KB
  __shared__ _Float16 plds[8][16 * 32];  // per-wave P staging (C->A layout)
  const int t    = threadIdx.x;
  const int lane = t & 31;
  const int wave = t >> 5;
  const int hf   = lane >> 4;
  const int l16  = lane & 15;
  const int h    = blockIdx.x >> 5;           // 32 blocks per head
  const int qb   = (blockIdx.x & 31) * 128;   // block query base
  const int q0   = qb + wave * 16;            // this wave's query tile
  _Float16* pbuf = plds[wave];

  const _Float16* qrow = qh + (size_t)(q0 + l16) * DMODEL + h * DK;
  const v16h aq0 = load_a_tile(qrow,  0, hf);
  const v16h aq1 = load_a_tile(qrow, 32, hf);

  // staging slots (one async b128 each for K and V per step)
  const int krow = t >> 3, kch = (t & 7) * 8;   // K: 32 rows x 128B
  _Float16* ksdst0 = Ksm[0] + krow * 64 + kch;
  _Float16* ksdst1 = Ksm[1] + krow * 64 + kch;
  const _Float16* kgsrc = kh + (size_t)krow * DMODEL + h * DK + kch;
  const int vrow = t >> 2, vch = (t & 3) * 8;   // V: 64 rows x 64B
  _Float16* vsdst0 = Vsm[0] + vrow * 32 + vch;
  _Float16* vsdst1 = Vsm[1] + vrow * 32 + vch;
  const _Float16* vgsrc = vt + (size_t)(h * DK + vrow) * SEQ + vch;

  float mrow[8], lrow[8];
  v8f acc[4] = {v8f{}, v8f{}, v8f{}, v8f{}};
#pragma unroll
  for (int r = 0; r < 8; ++r) { mrow[r] = -1e30f; lrow[r] = 0.0f; }

  const int jend = qb + 128;
  async_ld_b128(ksdst0, kgsrc);  // prologue fill of buffer 0
  async_ld_b128(vsdst0, vgsrc);
  for (int j0 = 0; j0 < jend; j0 += 32) {
    const int cur = (j0 >> 5) & 1;
    const _Float16* Kc = Ksm[cur];
    const _Float16* Vc = Vsm[cur];
    if (j0 + 32 < jend) {  // prefetch next K/V tiles into the other buffer
      async_ld_b128(cur ? ksdst0 : ksdst1, kgsrc + (size_t)(j0 + 32) * DMODEL);
      async_ld_b128(cur ? vsdst0 : vsdst1, vgsrc + (j0 + 32));
      wait_async2();  // oldest pair (current buffers) is resident
    } else {
      wait_async0();
    }
    __syncthreads();

    if (j0 < q0 + 16) {  // wave-uniform: EXEC stays all-ones for WMMA
      // ---- S = (Q K^T) * 1/sqrt(dk), 32 keys = two 16-key subtiles
      v8f s0 = {}, s1 = {};
      s0 = wmma_f16(aq0, load_b16(Kc + (l16 +  0) * 64 +  0 + hf * 16), s0);
      s0 = wmma_f16(aq1, load_b16(Kc + (l16 +  0) * 64 + 32 + hf * 16), s0);
      s1 = wmma_f16(aq0, load_b16(Kc + (l16 + 16) * 64 +  0 + hf * 16), s1);
      s1 = wmma_f16(aq1, load_b16(Kc + (l16 + 16) * 64 + 32 + hf * 16), s1);

      // ---- causal mask + online softmax (row M=r lanes 0-15, M=r+8 lanes 16-31)
      const int nA = j0 + l16;
      const int nB = j0 + 16 + l16;
#pragma unroll
      for (int r = 0; r < 8; ++r) {
        const int m_abs = q0 + r + 8 * hf;
        float v0 = s0[r] * 0.125f; if (nA > m_abs) v0 = -10000.0f;
        float v1 = s1[r] * 0.125f; if (nB > m_abs) v1 = -10000.0f;
        float mx = fmaxf(v0, v1);
        mx = fmaxf(mx, __shfl_xor(mx, 1, 32));
        mx = fmaxf(mx, __shfl_xor(mx, 2, 32));
        mx = fmaxf(mx, __shfl_xor(mx, 4, 32));
        mx = fmaxf(mx, __shfl_xor(mx, 8, 32));
        const float mnew = fmaxf(mrow[r], mx);
        const float sc = __expf(mrow[r] - mnew);
        mrow[r] = mnew;
        const float p0 = __expf(v0 - mnew);
        const float p1 = __expf(v1 - mnew);
        float rs = p0 + p1;
        rs += __shfl_xor(rs, 1, 32);
        rs += __shfl_xor(rs, 2, 32);
        rs += __shfl_xor(rs, 4, 32);
        rs += __shfl_xor(rs, 8, 32);
        lrow[r] = lrow[r] * sc + rs;
#pragma unroll
        for (int i = 0; i < 4; ++i) acc[i][r] *= sc;
        const int ml = r + 8 * hf;
        pbuf[ml * 32 + l16]      = (_Float16)p0;
        pbuf[ml * 32 + 16 + l16] = (_Float16)p1;
      }

      // ---- re-read P in A layout (intra-wave DS ops are in-order)
      const _Float16* prow = pbuf + l16 * 32;
      const v8h plo = *reinterpret_cast<const v8h*>(prow + hf * 16);
      const v8h phi = *reinterpret_cast<const v8h*>(prow + hf * 16 + 8);
      const v16h ap = __builtin_shufflevector(
          plo, phi, 0,1,2,3,4,5,6,7,8,9,10,11,12,13,14,15);

      // ---- acc += P @ V (Vsm rows = dk, contiguous local-key runs)
#pragma unroll
      for (int i = 0; i < 4; ++i) {
        v16h bv = load_b16(Vc + (i * 16 + l16) * 32 + hf * 16);
        acc[i] = wmma_f16(ap, bv, acc[i]);
      }
    }
    __syncthreads();  // all reads of 'cur' done before its next overwrite
  }

  // ---- normalize and emit context (fp16, [S, D] with head offset)
#pragma unroll
  for (int r = 0; r < 8; ++r) {
    const float inv = 1.0f / lrow[r];
    const int m = q0 + r + 8 * hf;
    _Float16* orow = ctxh + (size_t)m * DMODEL + h * DK;
#pragma unroll
    for (int i = 0; i < 4; ++i) orow[i * 16 + l16] = (_Float16)(acc[i][r] * inv);
  }
}

// ---- Output projection: out = ctx @ wo^T (fp32 out), LDS-staged B --------
__global__ __launch_bounds__(256) void out_gemm_kernel(
    const _Float16* __restrict__ ctxh, const _Float16* __restrict__ woh,
    float* __restrict__ out) {
  __shared__ _Float16 bsm[2][64 * 32];
  const int t    = threadIdx.x;
  const int lane = t & 31;
  const int wave = t >> 5;
  const int hf   = lane >> 4;
  const int l16  = lane & 15;
  const int m0 = blockIdx.x * 128 + wave * 16;
  const int n0 = blockIdx.y * 64;

  const _Float16* arow = ctxh + (size_t)(m0 + l16) * DMODEL;
  const int srow = t >> 2, sch = (t & 3) * 8;
  _Float16* sdst0 = bsm[0] + srow * 32 + sch;
  _Float16* sdst1 = bsm[1] + srow * 32 + sch;
  const _Float16* gsrc = woh + (size_t)(n0 + srow) * DMODEL + sch;

  v8f acc0 = {}, acc1 = {}, acc2 = {}, acc3 = {};
  async_ld_b128(sdst0, gsrc);
  for (int k0 = 0; k0 < DMODEL; k0 += 32) {
    const int cur = (k0 >> 5) & 1;
    const _Float16* bc = bsm[cur];
    if (k0 + 32 < DMODEL) {
      async_ld_b128(cur ? sdst0 : sdst1, gsrc + k0 + 32);
      wait_async1();
    } else {
      wait_async0();
    }
    __syncthreads();
    v16h a = load_a_tile(arow, k0, hf);
    acc0 = wmma_f16(a, load_b16(bc + (l16 +  0) * 32 + hf * 16), acc0);
    acc1 = wmma_f16(a, load_b16(bc + (l16 + 16) * 32 + hf * 16), acc1);
    acc2 = wmma_f16(a, load_b16(bc + (l16 + 32) * 32 + hf * 16), acc2);
    acc3 = wmma_f16(a, load_b16(bc + (l16 + 48) * 32 + hf * 16), acc3);
    __syncthreads();
  }
#pragma unroll
  for (int r = 0; r < 8; ++r) {
    const int m = m0 + r + 8 * hf;
    float* orow = out + (size_t)m * DMODEL + n0 + l16;
    orow[0]  = acc0[r];
    orow[16] = acc1[r];
    orow[32] = acc2[r];
    orow[48] = acc3[r];
  }
}

extern "C" void kernel_launch(void* const* d_in, const int* in_sizes, int n_in,
                              void* d_out, int out_size, void* d_ws, size_t ws_size,
                              hipStream_t stream) {
  const float* x  = (const float*)d_in[0];
  const float* wq = (const float*)d_in[1];
  const float* wk = (const float*)d_in[2];
  const float* wv = (const float*)d_in[3];
  const float* wo = (const float*)d_in[4];

  char* ws = (char*)d_ws;
  const size_t MB = 1u << 20;
  _Float16* xh   = (_Float16*)(ws + 0 * MB);   // 8 MB  [4096,1024]
  _Float16* wqh  = (_Float16*)(ws + 8 * MB);   // 2 MB
  _Float16* wkh  = (_Float16*)(ws + 10 * MB);  // 2 MB
  _Float16* wvh  = (_Float16*)(ws + 12 * MB);  // 2 MB
  _Float16* woh  = (_Float16*)(ws + 14 * MB);  // 2 MB
  _Float16* qh   = (_Float16*)(ws + 16 * MB);  // 8 MB  [4096,1024]
  _Float16* kh   = (_Float16*)(ws + 24 * MB);  // 8 MB  [4096,1024]
  _Float16* vt   = (_Float16*)(ws + 32 * MB);  // 8 MB  [1024,4096] (V^T)
  _Float16* ctxh = (_Float16*)(ws + 40 * MB);  // 8 MB  [4096,1024]

  const int nx = SEQ * DMODEL;
  const int nw = DMODEL * DMODEL;
  cvt_f32_f16_kernel<<<(nx + 255) / 256, 256, 0, stream>>>(x, xh, nx);
  cvt_f32_f16_kernel<<<(nw + 255) / 256, 256, 0, stream>>>(wq, wqh, nw);
  cvt_f32_f16_kernel<<<(nw + 255) / 256, 256, 0, stream>>>(wk, wkh, nw);
  cvt_f32_f16_kernel<<<(nw + 255) / 256, 256, 0, stream>>>(wv, wvh, nw);
  cvt_f32_f16_kernel<<<(nw + 255) / 256, 256, 0, stream>>>(wo, woh, nw);

  qkv_gemm_kernel<<<dim3(32, 16, 3), 256, 0, stream>>>(xh, wqh, wkh, wvh,
                                                       qh, kh, vt);
  attn_kernel<<<dim3(512), 256, 0, stream>>>(qh, kh, vt, ctxh);
  out_gemm_kernel<<<dim3(32, 16, 1), 256, 0, stream>>>(ctxh, woh,
                                                       (float*)d_out);
}
